// CVM_VIGOR_69784628625682
// MI455X (gfx1250) — compile-verified, matrix operand order checked by reading, not run on hardware
//
#include <hip/hip_runtime.h>
#include <hip/hip_bf16.h>

typedef __attribute__((ext_vector_type(16))) _Float16 v16h;
typedef __attribute__((ext_vector_type(8)))  _Float16 v8h;
typedef __attribute__((ext_vector_type(8)))  float    v8f;

#define BATCH   8
#define CDIM    64
#define HH      128
#define WW      128
#define NPIX    (HH * WW)          // 16384
#define NHEADS  8
#define NPOINTS 4
#define HDIM    8
#define BNROWS  (BATCH * NPIX)     // 131072

__device__ __forceinline__ v16h cat16(v8h lo, v8h hi) {
  return __builtin_shufflevector(lo, hi, 0, 1, 2, 3, 4, 5, 6, 7,
                                         8, 9, 10, 11, 12, 13, 14, 15);
}

// Load A/B fragment pair for K=64 directly from a row-major 64-half row.
// CDNA5 16-bit fragment layout: lane holds row (lane&15), khalf = lane>>4,
// element e -> k = e + 8*((e>>3) + khalf)  ==> two contiguous 8-half chunks.
__device__ __forceinline__ void load_frags(const _Float16* __restrict__ rowp,
                                           int khalf, v16h& f0, v16h& f1) {
  const _Float16* p = rowp + khalf * 8;
  v8h lo0 = *(const v8h*)(p);
  v8h hi0 = *(const v8h*)(p + 16);
  v8h lo1 = *(const v8h*)(p + 32);
  v8h hi1 = *(const v8h*)(p + 48);
  f0 = cat16(lo0, hi0);
  f1 = cat16(lo1, hi1);
}

// ---------------------------------------------------------------------------
// K0: pack weights to f16; concat [w_off;w_attn] -> 96x64, biases -> bcat(96)
// ---------------------------------------------------------------------------
__global__ void k_pack_weights(const float* __restrict__ w_off,  const float* __restrict__ b_off,
                               const float* __restrict__ w_attn, const float* __restrict__ b_attn,
                               const float* __restrict__ w_val,  const float* __restrict__ w_out,
                               _Float16* __restrict__ wcat16, _Float16* __restrict__ wval16,
                               _Float16* __restrict__ wout16, float* __restrict__ bcat) {
  int t = blockIdx.x * blockDim.x + threadIdx.x;
  if (t < 96 * 64)
    wcat16[t] = (_Float16)((t < 64 * 64) ? w_off[t] : w_attn[t - 64 * 64]);
  if (t < 64 * 64) {
    wval16[t] = (_Float16)w_val[t];
    wout16[t] = (_Float16)w_out[t];
  }
  if (t < 96)
    bcat[t] = (t < 64) ? b_off[t] : b_attn[t - 64];
}

// ---------------------------------------------------------------------------
// K1: q = query + sine PE (computed analytically), and value transposed to
// token-major; both stored f16 row-major (BN x 64) for WMMA A tiles.
// ---------------------------------------------------------------------------
__global__ void k_pack_qv(const float* __restrict__ query, const float* __restrict__ value,
                          _Float16* __restrict__ q16, _Float16* __restrict__ v16) {
  int t = blockIdx.x * blockDim.x + threadIdx.x;
  if (t >= BNROWS) return;
  int n = t & (NPIX - 1);
  int b = t >> 14;
  int wp = n & (WW - 1);
  int hp = n >> 7;
  float x = (float)(wp + 1);
  float y = (float)(hp + 1);
  const float kf = 0.28782313662425572f;   // ln(10000) / 32
  size_t qbase = (size_t)t * CDIM;
  size_t gbase = (size_t)b * CDIM * NPIX + (size_t)n;
#pragma unroll
  for (int i = 0; i < 16; ++i) {
    float f = __expf(-2.0f * (float)i * kf);
    float sx, cx, sy, cy;
    __sincosf(x * f, &sx, &cx);
    __sincosf(y * f, &sy, &cy);
    int c = 4 * i;
    q16[qbase + c + 0] = (_Float16)(query[gbase + (size_t)(c + 0) * NPIX] + sx);
    q16[qbase + c + 1] = (_Float16)(query[gbase + (size_t)(c + 1) * NPIX] + cx);
    q16[qbase + c + 2] = (_Float16)(query[gbase + (size_t)(c + 2) * NPIX] + sy);
    q16[qbase + c + 3] = (_Float16)(query[gbase + (size_t)(c + 3) * NPIX] + cy);
  }
#pragma unroll 8
  for (int c = 0; c < CDIM; ++c)
    v16[qbase + c] = (_Float16)value[gbase + (size_t)c * NPIX];
}

// ---------------------------------------------------------------------------
// WMMA GEMM: out(BN x OUTC) = A(BN x 64, f16) * W^T(OUTC x 64, f16) + bias.
// One wave per 64x16 output strip (four 16x16 tiles); K=64 via two v_wmma_f32_16x16x32_f16.
// Fragments loaded straight from global as b128 chunks (no LDS staging);
// weights are tiny and stay resident in L2 across blocks.
// ---------------------------------------------------------------------------
template <int OUTC>
__global__ void k_wmma_gemm(const _Float16* __restrict__ A, const _Float16* __restrict__ Wt,
                            const float* __restrict__ bias, float* __restrict__ out) {
  int tid   = threadIdx.x;
  int lane  = tid & 31;
  int wave  = tid >> 5;
  int m     = lane & 15;
  int khalf = lane >> 4;
  int col   = wave * 16 + m;
  float biasv = bias[col];

  v16h b0, b1;
  load_frags(Wt + (size_t)col * 64, khalf, b0, b1);

  int row00 = blockIdx.x * 64;
#pragma unroll
  for (int rt = 0; rt < 4; ++rt) {
    int row0 = row00 + rt * 16;
    const _Float16* ap = A + (size_t)(row0 + m) * 64;
    if (rt < 3) __builtin_prefetch(ap + 16 * 64, 0, 0);
    v16h a0, a1;
    load_frags(ap, khalf, a0, a1);

    v8f c;
#pragma unroll
    for (int r = 0; r < 8; ++r) c[r] = biasv;
    c = __builtin_amdgcn_wmma_f32_16x16x32_f16(false, a0, false, b0, (short)0, c, false, false);
    c = __builtin_amdgcn_wmma_f32_16x16x32_f16(false, a1, false, b1, (short)0, c, false, false);

    // D layout: lane (m,khalf), VGPR r -> row (r + 8*khalf), col = wave*16 + m.
    // Stores coalesce across lanes (consecutive cols).
#pragma unroll
    for (int r = 0; r < 8; ++r)
      out[(size_t)(row0 + r + 8 * khalf) * OUTC + col] = c[r];
  }
}

// ---------------------------------------------------------------------------
// K4: deformable sampling. One thread per (b, n, head): softmax over the 4
// point logits, zero-padded bilinear gather of 8-channel head slices.
// ---------------------------------------------------------------------------
__global__ void k_deform_sample(const float* __restrict__ offattn,
                                const float* __restrict__ vproj,
                                _Float16* __restrict__ samp) {
  int t = blockIdx.x * blockDim.x + threadIdx.x;
  if (t >= BNROWS * NHEADS) return;
  int head = t & (NHEADS - 1);
  int bn   = t >> 3;
  int n    = bn & (NPIX - 1);
  int b    = bn >> 14;
  int wp   = n & (WW - 1);
  int hp   = n >> 7;
  float refx = (float)wp * (1.0f / (float)(WW - 1));
  float refy = (float)hp * (1.0f / (float)(HH - 1));

  const float* oa = offattn + (size_t)bn * 96;
  float l0 = oa[64 + head * 4 + 0];
  float l1 = oa[64 + head * 4 + 1];
  float l2 = oa[64 + head * 4 + 2];
  float l3 = oa[64 + head * 4 + 3];
  float mx = fmaxf(fmaxf(l0, l1), fmaxf(l2, l3));
  float e0 = __expf(l0 - mx), e1 = __expf(l1 - mx);
  float e2 = __expf(l2 - mx), e3 = __expf(l3 - mx);
  float inv = 1.0f / (e0 + e1 + e2 + e3);
  float prob[4] = {e0 * inv, e1 * inv, e2 * inv, e3 * inv};

  const float* vb = vproj + (size_t)b * NPIX * CDIM + head * HDIM;
  float acc[HDIM];
#pragma unroll
  for (int d = 0; d < HDIM; ++d) acc[d] = 0.0f;

  auto corner = [&](int xi, int yi, float wc) {
    if (xi >= 0 && xi < WW && yi >= 0 && yi < HH) {
      const float4* src = (const float4*)(vb + (size_t)(yi * WW + xi) * CDIM);
      float4 s0 = src[0], s1 = src[1];
      acc[0] += wc * s0.x; acc[1] += wc * s0.y; acc[2] += wc * s0.z; acc[3] += wc * s0.w;
      acc[4] += wc * s1.x; acc[5] += wc * s1.y; acc[6] += wc * s1.z; acc[7] += wc * s1.w;
    }
  };

#pragma unroll
  for (int p = 0; p < NPOINTS; ++p) {
    float ox = oa[head * 8 + p * 2 + 0];
    float oy = oa[head * 8 + p * 2 + 1];
    // (ref + off/[W,H]) * [W,H] - 0.5
    float xx = refx * (float)WW + ox - 0.5f;
    float yy = refy * (float)HH + oy - 0.5f;
    float xf = floorf(xx), yf = floorf(yy);
    float wx = xx - xf, wy = yy - yf;
    int x0 = (int)xf, y0 = (int)yf;
    float aw  = prob[p];
    float w00 = aw * (1.0f - wx) * (1.0f - wy);
    float w01 = aw * wx * (1.0f - wy);
    float w10 = aw * (1.0f - wx) * wy;
    float w11 = aw * wx * wy;
    corner(x0,     y0,     w00);
    corner(x0 + 1, y0,     w01);
    corner(x0,     y0 + 1, w10);
    corner(x0 + 1, y0 + 1, w11);
  }

  _Float16* dst = samp + (size_t)bn * CDIM + head * HDIM;
#pragma unroll
  for (int d = 0; d < HDIM; ++d) dst[d] = (_Float16)acc[d];
}

// ---------------------------------------------------------------------------
// K5: output projection WMMA GEMM with transposing epilogue:
// out_map[b,c,n] = samp @ w_out^T + b_out ; writes (out_map+value, out_map).
// Accumulator tiles are transposed through a padded LDS slab so the two
// 67MB/2 output maps are written as coalesced float4 (b128) stores.
// ---------------------------------------------------------------------------
__global__ void k_wmma_gemm_out(const _Float16* __restrict__ A, const _Float16* __restrict__ Wt,
                                const float* __restrict__ bias, const float* __restrict__ value,
                                float* __restrict__ out_sum, float* __restrict__ out_map) {
  // stride 20 floats: 16B-aligned rows, conflict-free for the lane maps below
  __shared__ __attribute__((aligned(16))) float Ts[4][16][20];
  int tid   = threadIdx.x;
  int lane  = tid & 31;
  int wave  = tid >> 5;
  int m     = lane & 15;
  int khalf = lane >> 4;
  int col   = wave * 16 + m;
  float biasv = bias[col];

  v16h b0, b1;
  load_frags(Wt + (size_t)col * 64, khalf, b0, b1);

  int row00 = blockIdx.x * 64;
#pragma unroll
  for (int rt = 0; rt < 4; ++rt) {
    int row0 = row00 + rt * 16;
    const _Float16* ap = A + (size_t)(row0 + m) * 64;
    if (rt < 3) __builtin_prefetch(ap + 16 * 64, 0, 0);
    v16h a0, a1;
    load_frags(ap, khalf, a0, a1);

    v8f c;
#pragma unroll
    for (int r = 0; r < 8; ++r) c[r] = biasv;
    c = __builtin_amdgcn_wmma_f32_16x16x32_f16(false, a0, false, b0, (short)0, c, false, false);
    c = __builtin_amdgcn_wmma_f32_16x16x32_f16(false, a1, false, b1, (short)0, c, false, false);

    // Transpose tile: Ts[wave][c_in_tile][n_in_tile]
#pragma unroll
    for (int r = 0; r < 8; ++r)
      Ts[wave][m][r + 8 * khalf] = c[r];
    __syncthreads();

    // Each lane handles 8 consecutive n of one c-row -> float4 x2 stores
    int cc    = lane >> 1;
    int half  = lane & 1;
    int colg  = wave * 16 + cc;
    int b     = row0 >> 14;
    int nbase = (row0 & (NPIX - 1)) + half * 8;
    size_t idx = (size_t)b * CDIM * NPIX + (size_t)colg * NPIX + (size_t)nbase;

    float4 t0 = *(const float4*)&Ts[wave][cc][half * 8 + 0];
    float4 t1 = *(const float4*)&Ts[wave][cc][half * 8 + 4];
    *(float4*)(out_map + idx)     = t0;
    *(float4*)(out_map + idx + 4) = t1;

    float4 v0 = *(const float4*)(value + idx);
    float4 v1 = *(const float4*)(value + idx + 4);
    t0.x += v0.x; t0.y += v0.y; t0.z += v0.z; t0.w += v0.w;
    t1.x += v1.x; t1.y += v1.y; t1.z += v1.z; t1.w += v1.w;
    *(float4*)(out_sum + idx)     = t0;
    *(float4*)(out_sum + idx + 4) = t1;
    __syncthreads();
  }
}

// ---------------------------------------------------------------------------
extern "C" void kernel_launch(void* const* d_in, const int* in_sizes, int n_in,
                              void* d_out, int out_size, void* d_ws, size_t ws_size,
                              hipStream_t stream) {
  (void)in_sizes; (void)n_in; (void)out_size; (void)ws_size;
  const float* query  = (const float*)d_in[0];
  const float* value  = (const float*)d_in[1];
  const float* w_off  = (const float*)d_in[2];
  const float* b_off  = (const float*)d_in[3];
  const float* w_attn = (const float*)d_in[4];
  const float* b_attn = (const float*)d_in[5];
  const float* w_val  = (const float*)d_in[6];
  const float* b_val  = (const float*)d_in[7];
  const float* w_out  = (const float*)d_in[8];
  const float* b_out  = (const float*)d_in[9];

  char* ws = (char*)d_ws;
  const size_t SZ_ROW16 = (size_t)BNROWS * CDIM * sizeof(_Float16);   // 16.78 MB
  size_t off_q16     = 0;
  size_t off_v16     = off_q16 + SZ_ROW16;
  size_t off_wcat    = off_v16 + SZ_ROW16;
  size_t off_wval    = off_wcat + 96 * 64 * sizeof(_Float16);
  size_t off_wout    = off_wval + 64 * 64 * sizeof(_Float16);
  size_t off_bcat    = off_wout + 64 * 64 * sizeof(_Float16);
  size_t off_offattn = off_bcat + 512;
  size_t off_vproj   = off_offattn + (size_t)BNROWS * 96 * sizeof(float);
  size_t off_samp    = off_vproj + (size_t)BNROWS * CDIM * sizeof(float);

  _Float16* q16     = (_Float16*)(ws + off_q16);
  _Float16* v16     = (_Float16*)(ws + off_v16);
  _Float16* wcat16  = (_Float16*)(ws + off_wcat);
  _Float16* wval16  = (_Float16*)(ws + off_wval);
  _Float16* wout16  = (_Float16*)(ws + off_wout);
  float*    bcat    = (float*)(ws + off_bcat);
  float*    offattn = (float*)(ws + off_offattn);
  float*    vproj   = (float*)(ws + off_vproj);
  _Float16* samp    = (_Float16*)(ws + off_samp);

  float* out_sum = (float*)d_out;
  float* out_map = (float*)d_out + (size_t)BATCH * CDIM * NPIX;

  // K0: weight packing
  k_pack_weights<<<(96 * 64 + 255) / 256, 256, 0, stream>>>(
      w_off, b_off, w_attn, b_attn, w_val, w_out, wcat16, wval16, wout16, bcat);

  // K1: q = query + PE, value transpose (both -> f16 row-major)
  k_pack_qv<<<BNROWS / 256, 256, 0, stream>>>(query, value, q16, v16);

  // K2: q -> [offsets | attn logits]   (BN x 96)
  k_wmma_gemm<96><<<BNROWS / 64, 192, 0, stream>>>(q16, wcat16, bcat, offattn);

  // K3: value -> value projection      (BN x 64)
  k_wmma_gemm<64><<<BNROWS / 64, 128, 0, stream>>>(v16, wval16, b_val, vproj);

  // K4: softmax + bilinear deformable sampling -> samp (BN x 64, f16)
  k_deform_sample<<<(BNROWS * NHEADS) / 256, 256, 0, stream>>>(offattn, vproj, samp);

  // K5: output projection + epilogue (LDS transpose, +value, dual b128 store)
  k_wmma_gemm_out<<<BNROWS / 64, 128, 0, stream>>>(samp, wout16, b_out, value, out_sum, out_map);
}
